// AMLGraphSAGE_56435870269575
// MI455X (gfx1250) — compile-verified, matrix operand order checked by reading, not run on hardware
//
#include <hip/hip_runtime.h>
#include <hip/hip_bf16.h>
#include <math.h>

typedef __attribute__((ext_vector_type(16))) _Float16     v16h;
typedef __attribute__((ext_vector_type(8)))  float        v8f;
typedef __attribute__((ext_vector_type(4)))  unsigned int u32x4;
typedef __attribute__((ext_vector_type(8)))  int          i32x8;
typedef __attribute__((ext_vector_type(4)))  int          i32x4;

#define NN 100000
#define NE 2000000
// 100000 = 6250 * 16 exactly -> clean 16-row tiling
#define NTILES 6250

#if __has_builtin(__builtin_amdgcn_tensor_load_to_lds) && __has_builtin(__builtin_amdgcn_s_wait_tensorcnt)
#define USE_TDM 1
#else
#define USE_TDM 0
#endif

// ---------- Tensor Data Mover: 1-D contiguous f16 tile, global -> LDS ----------
// D# per CDNA5 ISA ch.8: group0 = {count/type/lds_addr/global_addr},
// group1 = {data_size, tensor_dim0/1, tile_dim0..2, strides}, groups 2/3 zero (<=2D).
#if USE_TDM
__device__ __forceinline__ void tdm_load_f16_1d(unsigned lds_off, const _Float16* gsrc,
                                                unsigned n_half /* < 65536, mult of 2 */) {
    unsigned long long ga = (unsigned long long)(uintptr_t)gsrc;
    u32x4 g0;
    g0[0] = 1u;                                            // count=1 valid descriptor
    g0[1] = lds_off;                                       // lds_addr (bytes)
    g0[2] = (unsigned)(ga & 0xFFFFFFFFu);                  // global_addr[31:0]
    g0[3] = (unsigned)((ga >> 32) & 0x1FFFFFFu) | (2u << 30); // global_addr[56:32] | type=2
    i32x8 g1 = {};
    g1[0] = (int)(1u << 16);                               // data_size=1 -> 2-byte elements
    g1[1] = (int)((n_half & 0xFFFFu) << 16);               // tensor_dim0[15:0]
    g1[2] = (int)(((n_half >> 16) & 0xFFFFu) | (1u << 16));// tensor_dim0[31:16] | tensor_dim1=1
    g1[3] = (int)((n_half & 0xFFFFu) << 16);               // tile_dim0 (16-bit)
    g1[4] = 0;                                             // tile_dim1=0 (1-D), tile_dim2=0
    g1[5] = (int)n_half;                                   // tensor_dim0_stride[31:0]
    g1[6] = 0;
    g1[7] = 0;
    i32x4 gz = {};
#if defined(__clang_major__) && (__clang_major__ >= 23)
    i32x8 gz8 = {};
    __builtin_amdgcn_tensor_load_to_lds(g0, g1, gz, gz, gz8, 0);
#else
    __builtin_amdgcn_tensor_load_to_lds(g0, g1, gz, gz, 0);
#endif
}
#endif

// ---------- WMMA fragment helpers (wave32, 16x16x32 f16, ISA 7.12.2) ----------
// 16-bit A 16x32: lanes 0-15 row M=lane hold K 0..7 (v0..3) and 16..23 (v4..7);
// lanes 16-31 row M=lane-16 hold K 8..15 and 24..31. B (32x16) mirrors with N<->M.
__device__ __forceinline__ int wmma_kmap(int i, int lane) {
    return (i & 7) | ((i >> 3) << 4) | ((lane >> 4) << 3);
}

// A fragment from row-major [16][ld] f16 tile in LDS (contiguous 8-half groups -> ds_load_b128)
__device__ __forceinline__ v16h load_a_lds(const _Float16* tile, int ld, int lane, int kbase) {
    v16h a;
    const int m = lane & 15;
#pragma unroll
    for (int i = 0; i < 16; ++i) a[i] = tile[m * ld + kbase + wmma_kmap(i, lane)];
    return a;
}

// B fragment from TRANSPOSED weights WT[n][k] (ldk = K): identical contiguous pattern
__device__ __forceinline__ v16h load_bT(const _Float16* wT, int ldk, int lane, int kbase, int nbase) {
    v16h b;
    const int n = nbase + (lane & 15);
#pragma unroll
    for (int i = 0; i < 16; ++i) b[i] = wT[n * ldk + kbase + wmma_kmap(i, lane)];
    return b;
}

__device__ __forceinline__ v8f wmma_f16(v16h a, v16h b, v8f c) {
    return __builtin_amdgcn_wmma_f32_16x16x32_f16(false, a, false, b, (short)0, c, false, false);
}

// ---------- utility kernels ----------
__global__ void fill_zero_kernel(float* p, unsigned n) {
    unsigned t = blockIdx.x * blockDim.x + threadIdx.x;
    if (t < n) p[t] = 0.0f;
}

// transpose-convert: W [K][N] f32 row-major -> WT [N][K] f16
__global__ void cvt_transpose_kernel(const float* __restrict__ s, _Float16* __restrict__ d,
                                     int K, int logN) {
    int t = blockIdx.x * blockDim.x + threadIdx.x;
    int total = K << logN;
    if (t >= total) return;
    int k = t >> logN;
    int n = t & ((1 << logN) - 1);
    d[n * K + k] = (_Float16)s[t];
}

__global__ void degree_kernel(const int* __restrict__ dst, float* cnt, int e_count) {
    int e = blockIdx.x * blockDim.x + threadIdx.x;
    if (e < e_count) atomicAdd(&cnt[dst[e]], 1.0f);
}

// scatter-add rows of h (F = 1<<logF floats) into ssum[dst]; one thread per (edge, feature)
__global__ void scatter_add_kernel(const float* __restrict__ h,
                                   const int* __restrict__ src,
                                   const int* __restrict__ dst,
                                   float* ssum, int e_count, int logF) {
    unsigned t = blockIdx.x * blockDim.x + threadIdx.x;
    unsigned total = (unsigned)e_count << logF;
    if (t >= total) return;
    unsigned e = t >> logF;
    unsigned f = t & ((1u << logF) - 1u);
    int s = src[e], d = dst[e];
    if (f == 0 && e + 64u < (unsigned)e_count) {
        // latency-hiding speculative prefetch of a future gathered row -> global_prefetch_b8
        __builtin_prefetch(&h[(size_t)src[e + 64u] << logF], 0, 0);
    }
    atomicAdd(&ssum[((size_t)d << logF) + f], h[((size_t)s << logF) + f]);
}

// ---------- h0 = x @ W0 + b0   ([16,32] x [32,64] per tile, one WMMA per n-tile) ----------
__global__ void __launch_bounds__(128) lin0_kernel(const float* __restrict__ x,
                                                   const _Float16* __restrict__ w0T, // [64][32]
                                                   const float* __restrict__ b0,
                                                   float* __restrict__ h0) {
    __shared__ _Float16 ax[16 * 32];
    __shared__ _Float16 sw0[64 * 32];
    const int tile = blockIdx.x;
    const int t = threadIdx.x;
    const int lane = t & 31, wave = t >> 5;   // wave = n-tile (N=64 -> 4 tiles)
#if USE_TDM
    if (wave == 0) {
        tdm_load_f16_1d((unsigned)(uintptr_t)(void*)sw0, w0T, 64 * 32);
    }
#else
    for (int i = t; i < 64 * 32; i += 128) sw0[i] = w0T[i];
#endif
    for (int i = t; i < 16 * 32; i += 128) {
        int m = i >> 5, k = i & 31;
        ax[i] = (_Float16)x[(size_t)(tile * 16 + m) * 32 + k];
    }
#if USE_TDM
    if (wave == 0) __builtin_amdgcn_s_wait_tensorcnt(0);
#endif
    __syncthreads();
    v8f c = {};
    v16h a = load_a_lds(ax, 32, lane, 0);
    v16h b = load_bT(sw0, 32, lane, 0, wave * 16);
    c = wmma_f16(a, b, c);
    const int n  = wave * 16 + (lane & 15);
    const int m0 = (lane >> 4) * 8;           // C layout: VGPR r -> M = r + 8*(lane>=16)
    const float bias = b0[n];
#pragma unroll
    for (int r = 0; r < 8; ++r)
        h0[(size_t)(tile * 16 + m0 + r) * 64 + n] = c[r] + bias;
}

// ---------- fused SAGE layer: out = act( (ssum/max(cnt,1)) @ Wl + bl + hin @ Wr ) ----------
__global__ void __launch_bounds__(128) sage_kernel(const float* __restrict__ ssum,
                                                   const float* __restrict__ cnt,
                                                   const float* __restrict__ hin,
                                                   const _Float16* __restrict__ wlT, // [ND][KD]
                                                   const _Float16* __restrict__ wrT, // [ND][KD]
                                                   const float* __restrict__ bl,
                                                   float* __restrict__ out,
                                                   int logKD, int ND, int do_relu) {
    __shared__ _Float16 sa[16 * 128];   // aggregated (mean) tile, f16
    __shared__ _Float16 sh[16 * 128];   // self tile, f16
    __shared__ _Float16 swl[8192];      // lin_l weights, transposed, f16 (KD*ND == 8192)
    __shared__ _Float16 swr[8192];      // lin_r weights, transposed, f16
    __shared__ float inv[16];
    const int KD = 1 << logKD;
    const int tile = blockIdx.x;
    const int t = threadIdx.x;
    const int lane = t & 31, wave = t >> 5;
#if USE_TDM
    if (wave == 0) {
        tdm_load_f16_1d((unsigned)(uintptr_t)(void*)swl, wlT, 8192);
        tdm_load_f16_1d((unsigned)(uintptr_t)(void*)swr, wrT, 8192);
    }
#else
    for (int i = t; i < 8192; i += 128) { swl[i] = wlT[i]; swr[i] = wrT[i]; }
#endif
    if (t < 16) {
        float c = cnt[tile * 16 + t];
        inv[t] = 1.0f / fmaxf(c, 1.0f);
    }
    __syncthreads();   // inv[] visible
    for (int i = t; i < (16 << logKD); i += 128) {
        int m = i >> logKD, k = i & (KD - 1);
        size_t node = (size_t)(tile * 16 + m);
        sa[m * KD + k] = (_Float16)(ssum[(node << logKD) + k] * inv[m]);
        sh[m * KD + k] = (_Float16)(hin[(node << logKD) + k]);
    }
#if USE_TDM
    if (wave == 0) __builtin_amdgcn_s_wait_tensorcnt(0);
#endif
    __syncthreads();
    const int ntiles = ND >> 4;
    for (int nt = wave; nt < ntiles; nt += 4) {
        v8f c = {};
        for (int kb = 0; kb < KD; kb += 32) {
            v16h aa  = load_a_lds(sa, KD, lane, kb);
            v16h bbl = load_bT(swl, KD, lane, kb, nt * 16);
            c = wmma_f16(aa, bbl, c);
            v16h ah  = load_a_lds(sh, KD, lane, kb);
            v16h bbr = load_bT(swr, KD, lane, kb, nt * 16);
            c = wmma_f16(ah, bbr, c);
        }
        const int n  = nt * 16 + (lane & 15);
        const int m0 = (lane >> 4) * 8;
        const float bias = bl[n];
#pragma unroll
        for (int r = 0; r < 8; ++r) {
            float v = c[r] + bias;
            if (do_relu) v = fmaxf(v, 0.0f);
            out[(size_t)(tile * 16 + m0 + r) * ND + n] = v;
        }
    }
}

// ---------- edge head: one wave32 per edge; raw = [h2[s],h2[d]] . Wp + bp ----------
__global__ void __launch_bounds__(256) edge_pred_kernel(const float* __restrict__ h2,
                                                        const int* __restrict__ src,
                                                        const int* __restrict__ dst,
                                                        const float* __restrict__ Wp,
                                                        const float* __restrict__ bp,
                                                        float* __restrict__ out_raw,
                                                        float* __restrict__ out_sig,
                                                        int e_count) {
    int gid = blockIdx.x * blockDim.x + threadIdx.x;
    int e = gid >> 5;
    int lane = gid & 31;
    if (e >= e_count) return;
    const float* hs = h2 + ((size_t)src[e] << 6);
    const float* hd = h2 + ((size_t)dst[e] << 6);
    float acc = hs[lane] * Wp[lane] + hs[lane + 32] * Wp[lane + 32]
              + hd[lane] * Wp[64 + lane] + hd[lane + 32] * Wp[96 + lane];
#pragma unroll
    for (int off = 16; off > 0; off >>= 1) acc += __shfl_xor(acc, off, 32);
    if (lane == 0) {
        float raw = acc + bp[0];
        out_raw[e] = raw;
        out_sig[e] = 1.0f / (1.0f + expf(-raw));
    }
}

extern "C" void kernel_launch(void* const* d_in, const int* in_sizes, int n_in,
                              void* d_out, int out_size, void* d_ws, size_t ws_size,
                              hipStream_t stream) {
    const float* x   = (const float*)d_in[0];
    const int*   ei  = (const int*)d_in[1];
    const float* W0  = (const float*)d_in[2];
    const float* b0  = (const float*)d_in[3];
    const float* W1l = (const float*)d_in[4];
    const float* b1l = (const float*)d_in[5];
    const float* W1r = (const float*)d_in[6];
    const float* W2l = (const float*)d_in[7];
    const float* b2l = (const float*)d_in[8];
    const float* W2r = (const float*)d_in[9];
    const float* Wp  = (const float*)d_in[10];
    const float* bp  = (const float*)d_in[11];
    const int* src = ei;            // edge_index[0]
    const int* dst = ei + NE;       // edge_index[1]

    // workspace layout (all deterministic, re-initialized every call)
    float* ws   = (float*)d_ws;
    float* cnt  = ws;                               // NN
    float* h0   = cnt  + NN;                        // NN*64
    float* h1   = h0   + (size_t)NN * 64;           // NN*128
    float* h2   = h1   + (size_t)NN * 128;          // NN*64
    float* ssum = h2   + (size_t)NN * 64;           // NN*128
    _Float16* W0t  = (_Float16*)(ssum + (size_t)NN * 128); // [64][32]
    _Float16* W1lt = W0t  + 64 * 32;                       // [128][64]
    _Float16* W1rt = W1lt + 128 * 64;                      // [128][64]
    _Float16* W2lt = W1rt + 128 * 64;                      // [64][128]
    _Float16* W2rt = W2lt + 64 * 128;                      // [64][128]

    float* out_raw = (float*)d_out;
    float* out_sig = out_raw + NE;

    // 1) f16 transposed weight staging (tiny)
    cvt_transpose_kernel<<<(32 * 64 + 255) / 256, 256, 0, stream>>>(W0,  W0t,  32,  6);
    cvt_transpose_kernel<<<(64 * 128 + 255) / 256, 256, 0, stream>>>(W1l, W1lt, 64,  7);
    cvt_transpose_kernel<<<(64 * 128 + 255) / 256, 256, 0, stream>>>(W1r, W1rt, 64,  7);
    cvt_transpose_kernel<<<(128 * 64 + 255) / 256, 256, 0, stream>>>(W2l, W2lt, 128, 6);
    cvt_transpose_kernel<<<(128 * 64 + 255) / 256, 256, 0, stream>>>(W2r, W2rt, 128, 6);

    // 2) zero degree counts + segment-sum buffer
    fill_zero_kernel<<<(NN + 255) / 256, 256, 0, stream>>>(cnt, NN);
    fill_zero_kernel<<<((unsigned)(NN * 128) + 255) / 256, 256, 0, stream>>>(ssum, (unsigned)NN * 128);

    // 3) degrees
    degree_kernel<<<(NE + 255) / 256, 256, 0, stream>>>(dst, cnt, NE);

    // 4) h0 = x @ W0 + b0   (WMMA + TDM weight staging)
    lin0_kernel<<<NTILES, 128, 0, stream>>>(x, W0t, b0, h0);

    // 5) scatter-mean(h0) -> ssum[:, :64]
    scatter_add_kernel<<<((unsigned)NE << 6) / 256, 256, 0, stream>>>(h0, src, dst, ssum, NE, 6);

    // 6) h1 = relu(SAGE1)  (WMMA, K=64, N=128)
    sage_kernel<<<NTILES, 128, 0, stream>>>(ssum, cnt, h0, W1lt, W1rt, b1l, h1, 6, 128, 1);

    // 7) re-zero ssum, scatter-mean(h1)
    fill_zero_kernel<<<((unsigned)(NN * 128) + 255) / 256, 256, 0, stream>>>(ssum, (unsigned)NN * 128);
    scatter_add_kernel<<<((unsigned)NE << 7) / 256, 256, 0, stream>>>(h1, src, dst, ssum, NE, 7);

    // 8) h2 = SAGE2 (no relu)  (WMMA, K=128, N=64)
    sage_kernel<<<NTILES, 128, 0, stream>>>(ssum, cnt, h1, W2lt, W2rt, b2l, h2, 7, 64, 0);

    // 9) edge prediction head: one wave per edge
    edge_pred_kernel<<<((unsigned)NE * 32 + 255) / 256, 256, 0, stream>>>(h2, src, dst, Wp, bp,
                                                                          out_raw, out_sig, NE);
}
